// HybridSGSSoftmaxModel_4183298146986
// MI455X (gfx1250) — matrix operand order repaired and checked by reading, not compiled, not run on hardware
//
#include <hip/hip_runtime.h>

// ---------------------------------------------------------------------------
// Problem constants (reference: B=128, S=512, V=30000, DS=64, DF=300)
// ---------------------------------------------------------------------------
#define BB   128
#define SS   512
#define DSD  64
#define DFD  300
#define DIN  600           // 2*DF
#define KP   608           // DIN padded to multiple of 32 (19 k-tiles)
#define NP1  608           // GEMM1 N padded (38 n-tiles)
#define NP2  304           // GEMM2 N padded (19 n-tiles)
#define KT   19            // k-tiles of 32
#define NT1  38            // n-tiles GEMM1
#define NT2  19            // n-tiles GEMM2
#define TOK  64            // tokens per block in the WMMA kernel (4 M-tiles)

typedef __attribute__((ext_vector_type(16))) _Float16 v16h;
typedef __attribute__((ext_vector_type(8)))  _Float16 v8h;
typedef __attribute__((ext_vector_type(8)))  float    v8f;

union FragH { v16h v; v8h h[2]; };

// A fragment (16x32 f16) from LDS, row-major [TOK][KP], rows mt*16..mt*16+15.
// Lane l: M = l&15. halves 0..15 -> K = kt*32 + (l>=16?16:0) + h
__device__ __forceinline__ v16h load_a_frag(const _Float16* base, int M, int kt, int hi) {
    FragH u;
    const _Float16* p = base + M * KP + kt * 32 + hi * 16;
    u.h[0] = *(const v8h*)p;
    u.h[1] = *(const v8h*)(p + 8);
    return u.v;
}

// B fragment (32x16 f16): row pointer pre-offset; contiguous 16 halves per lane.
__device__ __forceinline__ v16h load_b_row(const _Float16* prow, int kt) {
    FragH u;
    const _Float16* p = prow + kt * 32;
    u.h[0] = *(const v8h*)p;
    u.h[1] = *(const v8h*)(p + 8);
    return u.v;
}

__device__ __forceinline__ float gelu_exact(float x) {
    return 0.5f * x * (1.f + erff(x * 0.70710678118654752f));
}

// ---------------------------------------------------------------------------
// Weight conversion: src[K][N] f32 row-major -> dst[Np][Kp] f16 (transposed+pad)
// ---------------------------------------------------------------------------
__global__ void convert_wT_kernel(const float* __restrict__ src, _Float16* __restrict__ dst,
                                  int K, int N, int Kp, int Np) {
    int idx = blockIdx.x * blockDim.x + threadIdx.x;
    int total = Np * Kp;
    if (idx >= total) return;
    int n = idx / Kp, k = idx - n * Kp;
    float v = (n < N && k < K) ? src[(size_t)k * N + n] : 0.f;
    dst[idx] = (_Float16)v;
}

__global__ void zero_kernel(float* p, int n) {
    int i = blockIdx.x * blockDim.x + threadIdx.x;
    if (i < n) p[i] = 0.f;
}

// ---------------------------------------------------------------------------
// Stage 1 per (encode, batch): centroid, Gaussian render sgs, softmax weights
// ---------------------------------------------------------------------------
__global__ __launch_bounds__(256) void encode_stage1_kernel(
    const int* __restrict__ ids_a, const int* __restrict__ ids_b,
    const unsigned char* __restrict__ mask_a, const unsigned char* __restrict__ mask_b,
    const float* __restrict__ mu_tab, const float* __restrict__ logvar_tab,
    const float* __restrict__ alpha_tab, const float* __restrict__ feat_tab,
    const float* __restrict__ log_tau, const float* __restrict__ pos_mu,
    const float* __restrict__ pos_alpha,
    const float* __restrict__ wq, const float* __restrict__ bq,
    const float* __restrict__ wk, const float* __restrict__ bk,
    float* __restrict__ sgs_ws, float* __restrict__ wts_ws)
{
    __shared__ float sCent[DSD];
    __shared__ float sW[SS];
    __shared__ float sSc[SS];
    __shared__ float sRed[256];
    __shared__ float sSgs[DFD];
    __shared__ float sQ[DSD];
    __shared__ float sQk[DSD];
    __shared__ int   sIds[SS];
    __shared__ float sMisc[4];

    int gid = blockIdx.x;
    int e = gid >> 7, b = gid & 127;
    const int* ids = e ? ids_b : ids_a;
    const unsigned char* mask = e ? mask_b : mask_a;
    int tid = threadIdx.x;
    size_t eb = (size_t)e * BB + b;

    for (int s = tid; s < SS; s += 256) sIds[s] = ids[(size_t)b * SS + s];
    __syncthreads();
    float tau = __expf(log_tau[0]);

    // masked count
    float mcnt = 0.f;
    for (int s = tid; s < SS; s += 256) mcnt += mask[(size_t)b * SS + s] ? 1.f : 0.f;
    sRed[tid] = mcnt; __syncthreads();
    for (int st = 128; st > 0; st >>= 1) { if (tid < st) sRed[tid] += sRed[tid + st]; __syncthreads(); }
    float mfsum = sRed[0];
    __syncthreads();

    // centroid: 4 groups of 64 dims
    int d = tid & 63, g = tid >> 6;
    float cpart = 0.f;
    for (int s = g; s < SS; s += 4) {
        float mf = mask[(size_t)b * SS + s] ? 1.f : 0.f;
        int ps = s < 511 ? s : 511;
        cpart += mf * (mu_tab[(size_t)sIds[s] * DSD + d] + pos_mu[ps * DSD + d]);
    }
    sRed[tid] = cpart; __syncthreads();
    if (tid < DSD)
        sCent[tid] = (sRed[tid] + sRed[tid + 64] + sRed[tid + 128] + sRed[tid + 192]) / fmaxf(mfsum, 1.f);
    __syncthreads();

    // Gaussian kernel weights w = alpha' * K
    float wpart = 0.f;
    for (int s = tid; s < SS; s += 256) {
        int id = sIds[s];
        int ps = s < 511 ? s : 511;
        float mf = mask[(size_t)b * SS + s] ? 1.f : 0.f;
        float d2 = 0.f;
        for (int k = 0; k < DSD; k++) {
            float mu = mu_tab[(size_t)id * DSD + k] + pos_mu[ps * DSD + k];
            float df = mu - sCent[k];
            d2 += df * df * __expf(-logvar_tab[(size_t)id * DSD + k]);
        }
        float sig = 1.f / (1.f + __expf(-pos_alpha[ps]));
        float al = alpha_tab[id] * sig * mf;
        float Kk = __expf(-0.5f * d2 / tau) * mf;
        float w = al * Kk;
        sW[s] = w; wpart += w;
    }
    sRed[tid] = wpart; __syncthreads();
    for (int st = 128; st > 0; st >>= 1) { if (tid < st) sRed[tid] += sRed[tid + st]; __syncthreads(); }
    float denom = fmaxf(sRed[0], 1e-8f);
    __syncthreads();

    // sgs render
    for (int f = tid; f < DFD; f += 256) {
        float acc = 0.f;
        for (int s = 0; s < SS; s++) acc += sW[s] * feat_tab[(size_t)sIds[s] * DFD + f];
        float v = acc / denom;
        sSgs[f] = v;
        sgs_ws[eb * DFD + f] = v;
    }
    __syncthreads();

    // query = sgs @ wq + bq ; fold keys: score = mu . (wk @ q) + bk.q
    if (tid < DSD) {
        float q = bq[tid];
        for (int f2 = 0; f2 < DFD; f2++) q += sSgs[f2] * wq[(size_t)f2 * DSD + tid];
        sQ[tid] = q;
    }
    __syncthreads();
    if (tid < DSD) {
        float qk = 0.f;
        for (int ee = 0; ee < DSD; ee++) qk += wk[(size_t)tid * DSD + ee] * sQ[ee];
        sQk[tid] = qk;
    }
    if (tid == 0) {
        float cb = 0.f;
        for (int ee = 0; ee < DSD; ee++) cb += bk[ee] * sQ[ee];
        sMisc[1] = cb;
    }
    __syncthreads();

    // scores + softmax over S
    for (int s = tid; s < SS; s += 256) {
        int id = sIds[s];
        int ps = s < 511 ? s : 511;
        float sc = 0.f;
        for (int k = 0; k < DSD; k++)
            sc += (mu_tab[(size_t)id * DSD + k] + pos_mu[ps * DSD + k]) * sQk[k];
        sc = (sc + sMisc[1]) * 0.125f;   // 1/sqrt(64)
        sSc[s] = mask[(size_t)b * SS + s] ? sc : -1e30f;
    }
    __syncthreads();
    sRed[tid] = fmaxf(sSc[tid], sSc[tid + 256]); __syncthreads();
    for (int st = 128; st > 0; st >>= 1) { if (tid < st) sRed[tid] = fmaxf(sRed[tid], sRed[tid + st]); __syncthreads(); }
    float mx = sRed[0];
    __syncthreads();
    float epart = 0.f;
    for (int s = tid; s < SS; s += 256) { float ex = __expf(sSc[s] - mx); sSc[s] = ex; epart += ex; }
    sRed[tid] = epart; __syncthreads();
    for (int st = 128; st > 0; st >>= 1) { if (tid < st) sRed[tid] += sRed[tid + st]; __syncthreads(); }
    float inv = 1.f / sRed[0];
    for (int s = tid; s < SS; s += 256) wts_ws[eb * SS + s] = sSc[s] * inv;
}

// ---------------------------------------------------------------------------
// WMMA bridge MLP + weighted attention accumulation (the compute-bound core)
// 64 tokens per block (4 M-tiles sharing each B fragment), 8 waves,
// v_wmma_f32_16x16x32_f16. K-loop kept at unroll 1 with a manually
// double-buffered B fragment to bound register pressure (no acc shuffles).
// ---------------------------------------------------------------------------
__global__ __launch_bounds__(256) void mlp_attn_kernel(
    const int* __restrict__ ids_a, const int* __restrict__ ids_b,
    const float* __restrict__ feat_tab,
    const float* __restrict__ ln1_g, const float* __restrict__ ln1_b,
    const _Float16* __restrict__ W1h, const float* __restrict__ b1,
    const _Float16* __restrict__ W2h, const float* __restrict__ b2,
    const float* __restrict__ sgs, const float* __restrict__ wts,
    float* __restrict__ attn)
{
    __shared__ __align__(16) _Float16 sX[TOK * KP];   // 77824 B
    __shared__ __align__(16) _Float16 sH[TOK * KP];   // 77824 B
    __shared__ float sWt[TOK];
    __shared__ int   sId[TOK];
    __shared__ float sRed[256];
    __shared__ float sMV[TOK * 2];

    int bi = blockIdx.x;
    int e = bi >> 10;                 // 1024 tiles per encode
    int rem = bi & 1023;
    int b = rem >> 3;                 // 8 tiles per batch (512/64)
    int tile = rem & 7;
    int s0 = tile * TOK;
    const int* ids = e ? ids_b : ids_a;
    int tid = threadIdx.x;
    size_t eb = (size_t)e * BB + b;

    // ---- stage layernorm(concat(feat, sgs)) into LDS as f16 -------------
    int t = tid >> 2, j0 = tid & 3;   // 4 threads per token, 64 tokens
    if (j0 == 0) {
        sId[t] = ids[(size_t)b * SS + s0 + t];
        sWt[t] = wts[eb * SS + s0 + t];
    }
    __syncthreads();
    int idv = sId[t];
    const float* sg = sgs + eb * DFD;
    float s1 = 0.f, s2 = 0.f;
    for (int j = j0; j < DIN; j += 4) {
        float x = (j < DFD) ? feat_tab[(size_t)idv * DFD + j] : sg[j - DFD];
        s1 += x; s2 += x * x;
    }
    sRed[tid] = s1; __syncthreads();
    if (j0 == 0) { float a = 0; for (int k = 0; k < 4; k++) a += sRed[t * 4 + k]; sMV[t * 2] = a * (1.f / DIN); }
    __syncthreads();
    sRed[tid] = s2; __syncthreads();
    if (j0 == 0) {
        float a = 0; for (int k = 0; k < 4; k++) a += sRed[t * 4 + k];
        float m = sMV[t * 2];
        sMV[t * 2 + 1] = rsqrtf(a * (1.f / DIN) - m * m + 1e-5f);
    }
    __syncthreads();
    float m = sMV[t * 2], r = sMV[t * 2 + 1];
    for (int j = j0; j < KP; j += 4) {
        _Float16 hv = (_Float16)0.f;
        if (j < DIN) {
            float x = (j < DFD) ? feat_tab[(size_t)idv * DFD + j] : sg[j - DFD];
            hv = (_Float16)((x - m) * r * ln1_g[j] + ln1_b[j]);
        }
        sX[t * KP + j] = hv;
    }
    __syncthreads();

    int wave = tid >> 5, lane = tid & 31, ln = lane & 15, hi = lane >> 4;

    // ---- GEMM1: H = gelu(X @ W1 + b1), four M-tiles share each B frag ----
    for (int nt = wave; nt < NT1; nt += 8) {
        v8f acc0 = {0.f, 0.f, 0.f, 0.f, 0.f, 0.f, 0.f, 0.f};
        v8f acc1 = {0.f, 0.f, 0.f, 0.f, 0.f, 0.f, 0.f, 0.f};
        v8f acc2 = {0.f, 0.f, 0.f, 0.f, 0.f, 0.f, 0.f, 0.f};
        v8f acc3 = {0.f, 0.f, 0.f, 0.f, 0.f, 0.f, 0.f, 0.f};
        const _Float16* prow = W1h + (size_t)(nt * 16 + ln) * KP + hi * 16;
        v16h bm = load_b_row(prow, 0);
        #pragma unroll 1
        for (int kt = 0; kt < KT; ++kt) {
            // double-buffer: issue next B fragment load early (<=64B past row
            // end on the last tile, inside the 256B-padded workspace alloc)
            v16h bn = load_b_row(prow, kt + 1);
            __builtin_prefetch((const void*)(prow + (kt + 2) * 32), 0, 1);
            v16h a0 = load_a_frag(sX,           ln, kt, hi);
            v16h a1 = load_a_frag(sX + 16 * KP, ln, kt, hi);
            v16h a2 = load_a_frag(sX + 32 * KP, ln, kt, hi);
            v16h a3 = load_a_frag(sX + 48 * KP, ln, kt, hi);
            acc0 = __builtin_amdgcn_wmma_f32_16x16x32_f16(false, a0, false, bm, (short)0, acc0, false, false);
            acc1 = __builtin_amdgcn_wmma_f32_16x16x32_f16(false, a1, false, bm, (short)0, acc1, false, false);
            acc2 = __builtin_amdgcn_wmma_f32_16x16x32_f16(false, a2, false, bm, (short)0, acc2, false, false);
            acc3 = __builtin_amdgcn_wmma_f32_16x16x32_f16(false, a3, false, bm, (short)0, acc3, false, false);
            bm = bn;
        }
        int Nn = nt * 16 + ln;
        float bb = (Nn < DIN) ? b1[Nn] : 0.f;
        for (int v = 0; v < 8; ++v) {
            int M = v + hi * 8;
            sH[M * KP + Nn]        = (_Float16)gelu_exact(acc0[v] + bb);
            sH[(16 + M) * KP + Nn] = (_Float16)gelu_exact(acc1[v] + bb);
            sH[(32 + M) * KP + Nn] = (_Float16)gelu_exact(acc2[v] + bb);
            sH[(48 + M) * KP + Nn] = (_Float16)gelu_exact(acc3[v] + bb);
        }
    }
    __syncthreads();

    // ---- GEMM2: Y = H @ W2 + b2 ; attn += wt * (feat + Y) ----------------
    for (int nt = wave; nt < NT2; nt += 8) {
        v8f acc0 = {0.f, 0.f, 0.f, 0.f, 0.f, 0.f, 0.f, 0.f};
        v8f acc1 = {0.f, 0.f, 0.f, 0.f, 0.f, 0.f, 0.f, 0.f};
        v8f acc2 = {0.f, 0.f, 0.f, 0.f, 0.f, 0.f, 0.f, 0.f};
        v8f acc3 = {0.f, 0.f, 0.f, 0.f, 0.f, 0.f, 0.f, 0.f};
        const _Float16* prow = W2h + (size_t)(nt * 16 + ln) * KP + hi * 16;
        v16h bm = load_b_row(prow, 0);
        #pragma unroll 1
        for (int kt = 0; kt < KT; ++kt) {
            v16h bn = load_b_row(prow, kt + 1);
            __builtin_prefetch((const void*)(prow + (kt + 2) * 32), 0, 1);
            v16h a0 = load_a_frag(sH,           ln, kt, hi);
            v16h a1 = load_a_frag(sH + 16 * KP, ln, kt, hi);
            v16h a2 = load_a_frag(sH + 32 * KP, ln, kt, hi);
            v16h a3 = load_a_frag(sH + 48 * KP, ln, kt, hi);
            acc0 = __builtin_amdgcn_wmma_f32_16x16x32_f16(false, a0, false, bm, (short)0, acc0, false, false);
            acc1 = __builtin_amdgcn_wmma_f32_16x16x32_f16(false, a1, false, bm, (short)0, acc1, false, false);
            acc2 = __builtin_amdgcn_wmma_f32_16x16x32_f16(false, a2, false, bm, (short)0, acc2, false, false);
            acc3 = __builtin_amdgcn_wmma_f32_16x16x32_f16(false, a3, false, bm, (short)0, acc3, false, false);
            bm = bn;
        }
        int f = nt * 16 + ln;
        if (f < DFD) {
            float b2v = b2[f];
            float psum = 0.f;
            for (int v = 0; v < 8; ++v) {
                int M = v + hi * 8;
                psum += sWt[M]      * (acc0[v] + b2v + feat_tab[(size_t)sId[M]      * DFD + f]);
                psum += sWt[16 + M] * (acc1[v] + b2v + feat_tab[(size_t)sId[16 + M] * DFD + f]);
                psum += sWt[32 + M] * (acc2[v] + b2v + feat_tab[(size_t)sId[32 + M] * DFD + f]);
                psum += sWt[48 + M] * (acc3[v] + b2v + feat_tab[(size_t)sId[48 + M] * DFD + f]);
            }
            atomicAdd(&attn[eb * DFD + f], psum);
        }
    }
}

// ---------------------------------------------------------------------------
// Final MLP: final = sgs + MLP2(ln2(concat(sgs, attn)))  (256 rows, scalar)
// ---------------------------------------------------------------------------
__global__ __launch_bounds__(256) void final_mlp_kernel(
    const float* __restrict__ sgs, const float* __restrict__ attn,
    const float* __restrict__ ln2_g, const float* __restrict__ ln2_b,
    const float* __restrict__ aw1, const float* __restrict__ ab1,
    const float* __restrict__ aw2, const float* __restrict__ ab2,
    float* __restrict__ finalv)
{
    __shared__ float sx[DIN];
    __shared__ float sh[DIN];
    __shared__ float sRed[256];
    __shared__ float sMV[2];
    int gid = blockIdx.x;
    int e = gid >> 7, b = gid & 127;
    size_t eb = (size_t)e * BB + b;
    int tid = threadIdx.x;
    const float* sg = sgs + eb * DFD;
    const float* at = attn + eb * DFD;

    float s1 = 0.f, s2 = 0.f;
    for (int j = tid; j < DIN; j += 256) {
        float x = (j < DFD) ? sg[j] : at[j - DFD];
        sx[j] = x; s1 += x; s2 += x * x;
    }
    sRed[tid] = s1; __syncthreads();
    for (int st = 128; st > 0; st >>= 1) { if (tid < st) sRed[tid] += sRed[tid + st]; __syncthreads(); }
    if (tid == 0) sMV[0] = sRed[0] * (1.f / DIN);
    __syncthreads();
    sRed[tid] = s2; __syncthreads();
    for (int st = 128; st > 0; st >>= 1) { if (tid < st) sRed[tid] += sRed[tid + st]; __syncthreads(); }
    if (tid == 0) { float mm = sMV[0]; sMV[1] = rsqrtf(sRed[0] * (1.f / DIN) - mm * mm + 1e-5f); }
    __syncthreads();
    float mm = sMV[0], rr = sMV[1];
    for (int j = tid; j < DIN; j += 256) sx[j] = (sx[j] - mm) * rr * ln2_g[j] + ln2_b[j];
    __syncthreads();
    for (int j = tid; j < DIN; j += 256) {
        float a = ab1[j];
        for (int k = 0; k < DIN; k++) a += sx[k] * aw1[(size_t)k * DIN + j];
        sh[j] = gelu_exact(a);
    }
    __syncthreads();
    for (int f = tid; f < DFD; f += 256) {
        float y = ab2[f];
        for (int k = 0; k < DIN; k++) y += sh[k] * aw2[(size_t)k * DFD + f];
        finalv[eb * DFD + f] = sg[f] + y;
    }
}

// ---------------------------------------------------------------------------
// Cosine similarity * 5
// ---------------------------------------------------------------------------
__global__ void cosine_kernel(const float* __restrict__ finalv, float* __restrict__ out) {
    __shared__ float rA[64], rB[64], rC[64];
    int b = blockIdx.x, tid = threadIdx.x;
    const float* fa = finalv + (size_t)b * DFD;
    const float* fb = finalv + (size_t)(BB + b) * DFD;
    float da = 0.f, db = 0.f, dd = 0.f;
    for (int f = tid; f < DFD; f += 64) {
        float a = fa[f], c = fb[f];
        dd += a * c; da += a * a; db += c * c;
    }
    rA[tid] = da; rB[tid] = db; rC[tid] = dd; __syncthreads();
    for (int st = 32; st > 0; st >>= 1) {
        if (tid < st) { rA[tid] += rA[tid + st]; rB[tid] += rB[tid + st]; rC[tid] += rC[tid + st]; }
        __syncthreads();
    }
    if (tid == 0) {
        float na = fmaxf(sqrtf(rA[0]), 1e-8f);
        float nb = fmaxf(sqrtf(rB[0]), 1e-8f);
        out[b] = 5.f * rC[0] / (na * nb);
    }
}

// ---------------------------------------------------------------------------
extern "C" void kernel_launch(void* const* d_in, const int* in_sizes, int n_in,
                              void* d_out, int out_size, void* d_ws, size_t ws_size,
                              hipStream_t stream) {
    const int*            ids_a    = (const int*)d_in[0];
    const unsigned char*  mask_a   = (const unsigned char*)d_in[1];
    const int*            ids_b    = (const int*)d_in[2];
    const unsigned char*  mask_b   = (const unsigned char*)d_in[3];
    const float* mu_tab     = (const float*)d_in[4];
    const float* logvar_tab = (const float*)d_in[5];
    const float* alpha_tab  = (const float*)d_in[6];
    const float* feat_tab   = (const float*)d_in[7];
    const float* log_tau    = (const float*)d_in[8];
    const float* pos_mu     = (const float*)d_in[9];
    const float* pos_alpha  = (const float*)d_in[10];
    const float* ln1_g      = (const float*)d_in[11];
    const float* ln1_b      = (const float*)d_in[12];
    const float* w1         = (const float*)d_in[13];
    const float* b1         = (const float*)d_in[14];
    const float* w2         = (const float*)d_in[15];
    const float* b2         = (const float*)d_in[16];
    const float* wq         = (const float*)d_in[17];
    const float* bq         = (const float*)d_in[18];
    const float* wk         = (const float*)d_in[19];
    const float* bk         = (const float*)d_in[20];
    const float* ln2_g      = (const float*)d_in[21];
    const float* ln2_b      = (const float*)d_in[22];
    const float* aw1        = (const float*)d_in[23];
    const float* ab1        = (const float*)d_in[24];
    const float* aw2        = (const float*)d_in[25];
    const float* ab2        = (const float*)d_in[26];

    char* ws = (char*)d_ws;
    size_t off = 0;
    auto alloc = [&](size_t bytes) { size_t o = off; off += (bytes + 255) & ~(size_t)255; return o; };
    float*     sgs_ws  = (float*)(ws + alloc((size_t)2 * BB * DFD * 4));
    float*     wts_ws  = (float*)(ws + alloc((size_t)2 * BB * SS * 4));
    float*     attn_ws = (float*)(ws + alloc((size_t)2 * BB * DFD * 4));
    float*     fin_ws  = (float*)(ws + alloc((size_t)2 * BB * DFD * 4));
    _Float16*  W1h     = (_Float16*)(ws + alloc((size_t)NP1 * KP * 2));
    _Float16*  W2h     = (_Float16*)(ws + alloc((size_t)NP2 * KP * 2));

    // 1) weight conversion (transposed + padded f16)
    {
        int tot1 = NP1 * KP;
        convert_wT_kernel<<<(tot1 + 255) / 256, 256, 0, stream>>>(w1, W1h, DIN, DIN, KP, NP1);
        int tot2 = NP2 * KP;
        convert_wT_kernel<<<(tot2 + 255) / 256, 256, 0, stream>>>(w2, W2h, DIN, DFD, KP, NP2);
    }
    // 2) zero the attention accumulator (ws is poisoned by the harness)
    {
        int n = 2 * BB * DFD;
        zero_kernel<<<(n + 255) / 256, 256, 0, stream>>>(attn_ws, n);
    }
    // 3) stage 1: sgs + softmax weights, one block per (encode, batch)
    encode_stage1_kernel<<<2 * BB, 256, 0, stream>>>(
        ids_a, ids_b, mask_a, mask_b, mu_tab, logvar_tab, alpha_tab, feat_tab,
        log_tau, pos_mu, pos_alpha, wq, bq, wk, bk, sgs_ws, wts_ws);
    // 4) WMMA bridge MLP + attention accumulation (64 tokens per block)
    mlp_attn_kernel<<<2 * BB * (SS / TOK), 256, 0, stream>>>(
        ids_a, ids_b, feat_tab, ln1_g, ln1_b, W1h, b1, W2h, b2,
        sgs_ws, wts_ws, attn_ws);
    // 5) final MLP
    final_mlp_kernel<<<2 * BB, 256, 0, stream>>>(
        sgs_ws, attn_ws, ln2_g, ln2_b, aw1, ab1, aw2, ab2, fin_ws);
    // 6) cosine * 5
    cosine_kernel<<<BB, 64, 0, stream>>>(fin_ws, (float*)d_out);
}
